// CausalSelfAttention_30021821399686
// MI455X (gfx1250) — matrix operand order repaired and checked by reading, not compiled, not run on hardware
//
#include <hip/hip_runtime.h>
#include <hip/hip_bf16.h>
#include <math.h>

typedef __attribute__((ext_vector_type(16))) _Float16 v16h;
typedef __attribute__((ext_vector_type(8)))  float    v8f;
typedef __attribute__((ext_vector_type(4)))  unsigned int u32x4;
typedef __attribute__((ext_vector_type(8)))  int          i32x8;
typedef __attribute__((ext_vector_type(4)))  int          i32x4;

#define N_HEAD   16
#define HD       64
#define C_EMBD   1024
#define T_SEQ    2048
#define B_BATCH  2
#define WINDOW   256
#define BT       (B_BATCH * T_SEQ)   // 4096 rows total

// ---------------------------------------------------------------------------
// Tensor Data Mover: 2D tile (16-bit elements) global -> LDS.
// D# layout per CDNA5 ISA. One descriptor, no gather,
// no iterate/pad. tensor dims == tile dims (tiles always fully in-bounds).
// ---------------------------------------------------------------------------
static __device__ __forceinline__ void tdm_load_2d_f16(unsigned lds_byte_addr,
                                                       const void* gptr,
                                                       unsigned tile_d0,     // elems per row
                                                       unsigned tile_d1,     // rows
                                                       unsigned long long stride0) { // elems
  unsigned long long ga = (unsigned long long)gptr;
  u32x4 g0;
  g0[0] = 1u;                                             // count=1, user mode
  g0[1] = lds_byte_addr;                                  // lds_addr [63:32]
  g0[2] = (unsigned)(ga & 0xffffffffu);                   // global_addr[31:0]
  g0[3] = (unsigned)((ga >> 32) & 0x01ffffffu)            // global_addr[56:32]
        | 0x80000000u;                                    // type=2 ("image")
  i32x8 g1;
  g1[0] = 0x00010000;                                     // data_size=1 (2 bytes)
  g1[1] = (int)((tile_d0 & 0xffffu) << 16);               // tensor_dim0[15:0]
  g1[2] = (int)(((tile_d0 >> 16) & 0xffffu) |
                ((tile_d1 & 0xffffu) << 16));             // td0 hi | td1 lo
  g1[3] = (int)(((tile_d1 >> 16) & 0xffffu) |
                ((tile_d0 & 0xffffu) << 16));             // td1 hi | tile_dim0
  g1[4] = (int)(tile_d1 & 0xffffu);                       // tile_dim1, tile_dim2=0
  g1[5] = (int)(unsigned)(stride0 & 0xffffffffu);         // dim0_stride[31:0]
  g1[6] = (int)(unsigned)((stride0 >> 32) & 0xffffu);     // dim0_stride[47:32]
  g1[7] = 0;                                              // dim1_stride (unused, 2D)
  i32x4 z4 = {0, 0, 0, 0};
#if defined(__clang_major__) && (__clang_major__ >= 23)
  i32x8 z8 = {0, 0, 0, 0, 0, 0, 0, 0};
  __builtin_amdgcn_tensor_load_to_lds(g0, g1, z4, z4, z8, 0);
#else
  __builtin_amdgcn_tensor_load_to_lds(g0, g1, z4, z4, 0);
#endif
}

// ---------------------------------------------------------------------------
// WMMA fragment loaders (contiguous k = 16h + s order, wave32)
// ---------------------------------------------------------------------------

// A fragment 16x32 f16 from a global row-major matrix (stride lda).
static __device__ __forceinline__ v16h load_a16(const _Float16* A, int lda,
                                                int m0, int k0) {
  const int lane = threadIdx.x & 31;
  const int h = (lane >> 4) << 4;                 // 0 or 16
  const _Float16* p = A + (size_t)(m0 + (lane & 15)) * lda + k0 + h;
  v16h f;
  unsigned* u = reinterpret_cast<unsigned*>(&f);
#pragma unroll
  for (int j = 0; j < 4; ++j) {
    u[j]     = *reinterpret_cast<const unsigned*>(p + 2 * j);
    u[j + 4] = *reinterpret_cast<const unsigned*>(p + 8 + 2 * j);
  }
  return f;
}

// A fragment 16x32 f16 from an LDS tile laid out row-major [rows][32].
static __device__ __forceinline__ v16h lds_a_frag(const _Float16* As, int r0) {
  const int lane = threadIdx.x & 31;
  const _Float16* p = As + (r0 + (lane & 15)) * 32 + ((lane >> 4) << 4);
  v16h f;
  unsigned* u = reinterpret_cast<unsigned*>(&f);
#pragma unroll
  for (int j = 0; j < 4; ++j) {
    u[j]     = *reinterpret_cast<const unsigned*>(p + 2 * j);
    u[j + 4] = *reinterpret_cast<const unsigned*>(p + 8 + 2 * j);
  }
  return f;
}

// B fragment 32x16 f16 from an LDS pair-packed tile [16 pair-rows][256].
// lane: col = c0 + lane%16 ; VGPR j holds K = 16h+2j, 2j+1 -> pair-row 8h+j.
static __device__ __forceinline__ v16h lds_b_frag(const _Float16* Bs, int c0) {
  const int lane = threadIdx.x & 31;
  const _Float16* p = Bs + (((lane >> 4) << 3)) * 256 + (c0 + (lane & 15)) * 2;
  v16h f;
  unsigned* u = reinterpret_cast<unsigned*>(&f);
#pragma unroll
  for (int j = 0; j < 8; ++j)
    u[j] = *reinterpret_cast<const unsigned*>(p + j * 256);
  return f;
}

// B fragment for K^T: B[d][key] = Kmat[key][d] (row-major, stride ldk).
static __device__ __forceinline__ v16h load_bT_contig(const _Float16* Kmat, int ldk,
                                                      int key0, int d0) {
  const int lane = threadIdx.x & 31;
  const _Float16* p = Kmat + (size_t)(key0 + (lane & 15)) * ldk + d0 + ((lane >> 4) << 4);
  v16h f;
  unsigned* u = reinterpret_cast<unsigned*>(&f);
#pragma unroll
  for (int j = 0; j < 8; ++j)
    u[j] = *reinterpret_cast<const unsigned*>(p + 2 * j);
  return f;
}

// ---------------------------------------------------------------------------
// Precision conversion / layout kernels
// ---------------------------------------------------------------------------
__global__ void convert_f32_to_f16(const float* __restrict__ in,
                                   _Float16* __restrict__ out, int n) {
  int i = blockIdx.x * blockDim.x + threadIdx.x;
  if (i < n) out[i] = (_Float16)in[i];
}

// convert + pair-pack a [K x N] row-major weight: BP[(k/2)*2N + n*2 + (k&1)]
__global__ void convert_pack_b(const float* __restrict__ in,
                               _Float16* __restrict__ out, int K, int N) {
  long long i = (long long)blockIdx.x * blockDim.x + threadIdx.x;
  if (i >= (long long)K * N) return;
  int k = (int)(i / N), n = (int)(i % N);
  out[(size_t)(k >> 1) * N * 2 + (size_t)n * 2 + (k & 1)] = (_Float16)in[i];
}

// V slice of qkv [BT][3C] -> vT [B][H][HD][T]  (so P*V B-fragments are b32)
__global__ void transpose_v(const _Float16* __restrict__ qkv,
                            _Float16* __restrict__ vT) {
  int i = blockIdx.x * blockDim.x + threadIdx.x;       // over BT*C
  if (i >= BT * C_EMBD) return;
  const int bt = i / C_EMBD, c = i % C_EMBD;
  const int b = bt / T_SEQ, t = bt % T_SEQ;
  const int h = c / HD, d = c % HD;
  vT[((size_t)(b * N_HEAD + h) * HD + d) * T_SEQ + t] =
      qkv[(size_t)bt * (3 * C_EMBD) + 2 * C_EMBD + c];
}

// ---------------------------------------------------------------------------
// TDM-staged WMMA GEMM:  C[MxN] = A[MxK](f16 row-major) * BP (f16 pair-packed)
// 128x128 block tile, 8 waves each computing 32x64. Double-buffered LDS tiles
// filled by the Tensor Data Mover (wave 0 issues, TENSORcnt synchronizes).
// ---------------------------------------------------------------------------
template <typename OutT>
__global__ void __launch_bounds__(256)
wmma_gemm_tdm(const _Float16* __restrict__ A, const _Float16* __restrict__ BP,
              OutT* __restrict__ Cm, int M, int N, int K) {
  __shared__ _Float16 lA[2][128 * 32];   // [row][k]      8KB per buffer
  __shared__ _Float16 lB[2][16 * 256];   // [pairrow][2n] 8KB per buffer
  const int wave = threadIdx.x >> 5;
  const int lane = threadIdx.x & 31;
  const int m0 = blockIdx.y * 128;
  const int n0 = blockIdx.x * 128;
  const int wm = (wave & 3) * 32;        // wave rows within block tile
  const int wn = (wave >> 2) * 64;       // wave cols within block tile

  auto issue = [&](int buf, int k0) {
    if (wave == 0) {
      tdm_load_2d_f16((unsigned)reinterpret_cast<unsigned long long>(&lA[buf][0]),
                      A + (size_t)m0 * K + k0, 32u, 128u, (unsigned long long)K);
      tdm_load_2d_f16((unsigned)reinterpret_cast<unsigned long long>(&lB[buf][0]),
                      BP + (size_t)(k0 >> 1) * (2 * N) + (size_t)n0 * 2,
                      256u, 16u, (unsigned long long)(2 * N));
    }
  };

  issue(0, 0);
  v8f acc[2][4] = {};
  int buf = 0;
  for (int k0 = 0; k0 < K; k0 += 32, buf ^= 1) {
    const bool more = (k0 + 32) < K;
    if (more) issue(buf ^ 1, k0 + 32);
    if (wave == 0) {
      if (more) __builtin_amdgcn_s_wait_tensorcnt(2);   // newest 2 may fly
      else      __builtin_amdgcn_s_wait_tensorcnt(0);
    }
    __syncthreads();
    const _Float16* As = &lA[buf][0];
    const _Float16* Bs = &lB[buf][0];
    v16h a0 = lds_a_frag(As, wm);
    v16h a1 = lds_a_frag(As, wm + 16);
#pragma unroll
    for (int i = 0; i < 4; ++i) {
      v16h b = lds_b_frag(Bs, wn + 16 * i);
      acc[0][i] = __builtin_amdgcn_wmma_f32_16x16x32_f16(false, a0, false, b, (short)0, acc[0][i], false, false);
      acc[1][i] = __builtin_amdgcn_wmma_f32_16x16x32_f16(false, a1, false, b, (short)0, acc[1][i], false, false);
    }
    __syncthreads();                     // all waves done with buf before refill
  }

  const int col = lane & 15;
  const int rb  = (lane >> 4) << 3;
#pragma unroll
  for (int t = 0; t < 2; ++t)
#pragma unroll
    for (int i = 0; i < 4; ++i)
#pragma unroll
      for (int r = 0; r < 8; ++r)
        Cm[(size_t)(m0 + wm + t * 16 + rb + r) * N + n0 + wn + 16 * i + col] =
            (OutT)acc[t][i][r];
}

// ---------------------------------------------------------------------------
// Windowed causal attention, flash-style streaming softmax.
// One wave per (batch, head, 16-query tile); 32 keys (2 tiles) per iteration.
// ---------------------------------------------------------------------------
__global__ void __launch_bounds__(256)
attn_kernel(const _Float16* __restrict__ qkv, const _Float16* __restrict__ vT,
            _Float16* __restrict__ y) {
  __shared__ _Float16 pbuf[8][16 * 32];          // per-wave P tile [16 q][32 k]
  const int wave = threadIdx.x >> 5;
  const int lane = threadIdx.x & 31;
  const int task = blockIdx.x * 8 + wave;        // 4096 tasks total
  const int qt   = task & 127;                   // T/16 = 128 query tiles
  const int h    = (task >> 7) & 15;
  const int b    = task >> 11;
  const int q0   = qt * 16;
  const int ld   = 3 * C_EMBD;
  const _Float16* Qp  = qkv + (size_t)(b * T_SEQ) * ld + h * HD;
  const _Float16* Kp  = Qp + C_EMBD;
  const _Float16* VTh = vT + (size_t)(b * N_HEAD + h) * HD * T_SEQ;  // [d][t]

  const v16h qa0 = load_a16(Qp, ld, q0, 0);
  const v16h qa1 = load_a16(Qp, ld, q0, 32);

  v8f   yacc[4] = {};
  float rmax[8], rsum[8];
#pragma unroll
  for (int r = 0; r < 8; ++r) { rmax[r] = -INFINITY; rsum[r] = 0.0f; }

  const int col  = lane & 15;
  const int half = lane >> 4;
  int lo = q0 - (WINDOW - 1); if (lo < 0) lo = 0;
  const int kt_lo = lo >> 4;
  const int ntile = qt - kt_lo + 1;
  const int npair = (ntile + 1) >> 1;

  for (int it = 0; it < npair; ++it) {
    const int  kposA  = (kt_lo + 2 * it) * 16;
    const bool validB = (2 * it + 1) < ntile;
    const int  kposB  = validB ? kposA + 16 : kposA;   // safe duplicate if odd

    // prefetch next iteration's K rows and V^T columns (global_prefetch_b8)
    if (it + 1 < npair) {
      __builtin_prefetch(Kp + (size_t)(kposA + 32 + col) * ld, 0, 1);
      __builtin_prefetch(VTh + (size_t)col * T_SEQ + kposA + 32, 0, 1);
    }

    // ---- scores for both key tiles: 4 WMMAs ----
    v8f sA = {}, sB = {};
    {
      v16h bk = load_bT_contig(Kp, ld, kposA, 0);
      sA = __builtin_amdgcn_wmma_f32_16x16x32_f16(false, qa0, false, bk, (short)0, sA, false, false);
      bk = load_bT_contig(Kp, ld, kposA, 32);
      sA = __builtin_amdgcn_wmma_f32_16x16x32_f16(false, qa1, false, bk, (short)0, sA, false, false);
      bk = load_bT_contig(Kp, ld, kposB, 0);
      sB = __builtin_amdgcn_wmma_f32_16x16x32_f16(false, qa0, false, bk, (short)0, sB, false, false);
      bk = load_bT_contig(Kp, ld, kposB, 32);
      sB = __builtin_amdgcn_wmma_f32_16x16x32_f16(false, qa1, false, bk, (short)0, sB, false, false);
    }

    // ---- mask + streaming softmax stats ----
    float fac[8];
#pragma unroll
    for (int r = 0; r < 8; ++r) {
      const int qrow = q0 + (half << 3) + r;
      const int relA = qrow - (kposA + col);
      const int relB = qrow - (kposB + col);
      float svA = 0.125f * sA[r];                       // 1/sqrt(64)
      float svB = 0.125f * sB[r];
      if (!(relA >= 0 && relA < WINDOW)) svA = -INFINITY;
      if (!validB || !(relB >= 0 && relB < WINDOW)) svB = -INFINITY;
      float mt = fmaxf(svA, svB);
#pragma unroll
      for (int off = 1; off < 16; off <<= 1) mt = fmaxf(mt, __shfl_xor(mt, off));
      const float mnew = fmaxf(rmax[r], mt);
      const float pA = (svA > -INFINITY) ? __expf(svA - mnew) : 0.0f;
      const float pB = (svB > -INFINITY) ? __expf(svB - mnew) : 0.0f;
      float ps = pA + pB;
#pragma unroll
      for (int off = 1; off < 16; off <<= 1) ps += __shfl_xor(ps, off);
      fac[r]  = (rmax[r] > -INFINITY) ? __expf(rmax[r] - mnew) : 0.0f;
      rsum[r] = rsum[r] * fac[r] + ps;
      rmax[r] = mnew;
      const int m32 = ((half << 3) + r) * 32;
      pbuf[wave][m32 + col]      = (_Float16)pA;
      pbuf[wave][m32 + 16 + col] = (_Float16)pB;
    }

    // rescale running output by exp(m_old - m_new)
#pragma unroll
    for (int i = 0; i < 4; ++i)
#pragma unroll
      for (int r = 0; r < 8; ++r) yacc[i][r] *= fac[r];

    // ---- A = P (full K=32: keys of both tiles), from LDS ----
    v16h pa;
    {
      const int m  = col;                      // A row = lane%16
      const int hh = half << 4;
      unsigned* u = reinterpret_cast<unsigned*>(&pa);
#pragma unroll
      for (int j = 0; j < 4; ++j) {
        u[j]     = *reinterpret_cast<const unsigned*>(&pbuf[wave][m * 32 + hh + 2 * j]);
        u[j + 4] = *reinterpret_cast<const unsigned*>(&pbuf[wave][m * 32 + 8 + hh + 2 * j]);
      }
    }
    // ---- O += P * V : 4 full-K WMMAs; V^T gives contiguous key pairs ----
    const int kbase = half ? kposB : kposA;    // B VGPR j holds K = 16h + 2j
#pragma unroll
    for (int i = 0; i < 4; ++i) {
      v16h vb;
      unsigned* u = reinterpret_cast<unsigned*>(&vb);
      const _Float16* pv = VTh + (size_t)(i * 16 + col) * T_SEQ + kbase;
#pragma unroll
      for (int j = 0; j < 8; ++j)
        u[j] = *reinterpret_cast<const unsigned*>(pv + 2 * j);
      yacc[i] = __builtin_amdgcn_wmma_f32_16x16x32_f16(false, pa, false, vb, (short)0, yacc[i], false, false);
    }
  }

  // ---- normalize and store y tile (f16, row-major [BT x C]) ----
  _Float16* Yp = y + (size_t)(b * T_SEQ + q0) * C_EMBD + h * HD;
#pragma unroll
  for (int i = 0; i < 4; ++i)
#pragma unroll
    for (int r = 0; r < 8; ++r) {
      const float v = yacc[i][r] / rsum[r];
      Yp[(size_t)((half << 3) + r) * C_EMBD + i * 16 + col] = (_Float16)v;
    }
}

// ---------------------------------------------------------------------------
// Launch
// ---------------------------------------------------------------------------
extern "C" void kernel_launch(void* const* d_in, const int* in_sizes, int n_in,
                              void* d_out, int out_size, void* d_ws, size_t ws_size,
                              hipStream_t stream) {
  const float* x      = (const float*)d_in[0];   // [B,T,C]
  const float* w_attn = (const float*)d_in[1];   // [C, 3C]
  const float* w_proj = (const float*)d_in[2];   // [C, C]
  float* out = (float*)d_out;                    // [B,T,C] f32

  _Float16* ws    = (_Float16*)d_ws;
  _Float16* x16   = ws;                                        // BT*C
  _Float16* wa16  = x16  + (size_t)BT * C_EMBD;                // C*3C (packed)
  _Float16* wp16  = wa16 + (size_t)C_EMBD * 3 * C_EMBD;        // C*C  (packed)
  _Float16* qkv16 = wp16 + (size_t)C_EMBD * C_EMBD;            // BT*3C
  _Float16* y16   = qkv16 + (size_t)BT * 3 * C_EMBD;           // BT*C
  _Float16* vT16  = y16   + (size_t)BT * C_EMBD;               // BT*C (V^T)

  const int nx = BT * C_EMBD;
  const int na = C_EMBD * 3 * C_EMBD;
  const int np = C_EMBD * C_EMBD;
  convert_f32_to_f16<<<(nx + 255) / 256, 256, 0, stream>>>(x, x16, nx);
  convert_pack_b   <<<(na + 255) / 256, 256, 0, stream>>>(w_attn, wa16, C_EMBD, 3 * C_EMBD);
  convert_pack_b   <<<(np + 255) / 256, 256, 0, stream>>>(w_proj, wp16, C_EMBD, C_EMBD);

  // qkv = x @ w_attn : [4096 x 3072], K=1024
  dim3 g1(3 * C_EMBD / 128, BT / 128);
  wmma_gemm_tdm<_Float16><<<g1, 256, 0, stream>>>(x16, wa16, qkv16, BT, 3 * C_EMBD, C_EMBD);

  // V slice -> [B][H][d][T] for contiguous P*V fragments
  transpose_v<<<(nx + 255) / 256, 256, 0, stream>>>(qkv16, vT16);

  // windowed attention: 4096 waves = 512 blocks x 8 waves
  attn_kernel<<<(B_BATCH * N_HEAD * (T_SEQ / 16)) / 8, 256, 0, stream>>>(qkv16, vT16, y16);

  // out = y @ w_proj : [4096 x 1024], K=1024, f32 output
  dim3 g2(C_EMBD / 128, BT / 128);
  wmma_gemm_tdm<float><<<g2, 256, 0, stream>>>(y16, wp16, out, BT, C_EMBD, C_EMBD);
}
